// model1_59708635349629
// MI455X (gfx1250) — compile-verified, matrix operand order checked by reading, not run on hardware
//
#include <hip/hip_runtime.h>
#include <stdint.h>

#define BB 32
#define VV 16
#define TT 64
#define FF 64
#define HH 512
#define G4H 2048
#define NBINS 36
#define OUTN 4
#define NWG 64
#define NTHR 256
#define KT_ENC 18   /* (64+512)/32 k-tiles */
#define KT_SOFT 32  /* 1024/32 k-tiles */

typedef __attribute__((ext_vector_type(16))) __bf16 v16bf;
typedef __attribute__((ext_vector_type(8)))  float  v8f;
typedef uint16_t u16;

/* ---------------- workspace layout (bytes, all 256-aligned) ---------------- */
#define WS_CTR    0u                          /* 4B grid-barrier counter      */
#define WS_MAX    256u                        /* 64 * i32 per-step max slots  */
#define WS_SEQBF  512u                        /* B*V*T*F bf16   = 4,194,304   */
#define WS_XDEC   (WS_SEQBF + 4194304u)       /* B*V*F bf16     = 65,536      */
#define WS_WSW    (WS_XDEC  + 65536u)         /* V*128*18*512 bf16 = 37,748,736 */
#define WS_SOFTW  (WS_WSW   + 37748736u)      /* V*32*32*512 bf16 = 16,777,216 */
#define WS_BIAS   (WS_SOFTW + 16777216u)      /* V*4H f32       = 131,072     */
#define WS_WN     (WS_BIAS  + 131072u)        /* 64*B*V*V f32   = 2,097,152   */
#define WS_HBUF   (WS_WN    + 2097152u)       /* B*V*H bf16     = 524,288     */
#define WS_ENC    (WS_HBUF  + 524288u)        /* B*V*T*H bf16   = 33,554,432  */
#define WS_ZBUF   (WS_ENC   + 33554432u)      /* V*32*1024 bf16 = 1,048,576   */
#define WS_H2     (WS_ZBUF  + 1048576u)       /* 32*V*H f32     = 1,048,576   */

__device__ __forceinline__ u16 f2bf(float x) {
    uint32_t u = __float_as_uint(x);
    uint32_t r = u + 0x7fffu + ((u >> 16) & 1u);   /* round-nearest-even */
    return (u16)(r >> 16);
}
__device__ __forceinline__ float bf2f(u16 h) {
    return __uint_as_float(((uint32_t)h) << 16);
}
__device__ __forceinline__ float sigm(float x) { return 1.0f / (1.0f + __expf(-x)); }

/* ISA §7.12.2: 16-bit A/B fragment K index for element e, lane-half */
__device__ __forceinline__ int kmap(int e, int half) {
    int vg = e >> 1;
    return 2 * (vg & 3) + 16 * (vg >> 2) + 8 * half + (e & 1);
}
__device__ __forceinline__ v8f vsplat(float s) {
    v8f r;
#pragma unroll
    for (int i = 0; i < 8; ++i) r[i] = s;
    return r;
}

__device__ __forceinline__ void grid_barrier(unsigned* ctr, unsigned& nbar) {
    __syncthreads();
    ++nbar;
    if (threadIdx.x == 0) {
        __threadfence();
        __hip_atomic_fetch_add(ctr, 1u, __ATOMIC_RELEASE, __HIP_MEMORY_SCOPE_AGENT);
        while (__hip_atomic_load(ctr, __ATOMIC_ACQUIRE, __HIP_MEMORY_SCOPE_AGENT) <
               nbar * (unsigned)NWG)
            __builtin_amdgcn_s_sleep(8);
    }
    __syncthreads();
}

/* =================== init kernel 0: zero state =================== */
__global__ void k_zero(char* ws) {
    size_t gid = (size_t)blockIdx.x * blockDim.x + threadIdx.x;
    size_t stride = (size_t)gridDim.x * blockDim.x;
    if (gid == 0) *(unsigned*)(ws + WS_CTR) = 0u;
    if (gid < 64) ((int*)(ws + WS_MAX))[gid] = 0;
    u16* hbuf = (u16*)(ws + WS_HBUF);
    for (size_t i = gid; i < (size_t)BB * VV * HH; i += stride) hbuf[i] = 0;
}

/* =================== init kernel 1: convert / swizzle / hardwired w ========= */
__global__ void k_prep(char* ws,
                       const float* seq, const float* dist, const int* rb,
                       const int* cog, const float* mask, const float* maxv,
                       const float* minv, const float* W_ih, const float* W_hh,
                       const float* b_ih, const float* b_hh, const float* dom,
                       const float* soft_W) {
    size_t gid = (size_t)blockIdx.x * blockDim.x + threadIdx.x;
    size_t stride = (size_t)gridDim.x * blockDim.x;

    u16*   seq_bf = (u16*)(ws + WS_SEQBF);
    u16*   xdec   = (u16*)(ws + WS_XDEC);
    u16*   Wsw    = (u16*)(ws + WS_WSW);
    u16*   softWs = (u16*)(ws + WS_SOFTW);
    float* bias   = (float*)(ws + WS_BIAS);
    float* wn     = (float*)(ws + WS_WN);
    int*   maxi   = (int*)(ws + WS_MAX);

    /* A) sequence -> bf16 */
    for (size_t i = gid; i < (size_t)BB * VV * TT * FF; i += stride)
        seq_bf[i] = f2bf(seq[i]);

    /* B) decoder input: denormalized last step */
    for (size_t i = gid; i < (size_t)BB * VV * FF; i += stride) {
        int f = (int)(i & 63), v = (int)((i >> 6) & 15), b = (int)(i >> 10);
        float val = seq[(((size_t)(b * VV + v) * TT) + (TT - 1)) * FF + f] *
                        (maxv[b * FF + f] - minv[b * FF + f]) +
                    minv[b * FF + f];
        xdec[i] = f2bf(val);
    }

    /* C) combined bias */
    for (size_t i = gid; i < (size_t)VV * G4H; i += stride)
        bias[i] = b_ih[i] + b_hh[i];

    /* D) encoder weights -> fragment-swizzled bf16: [v][ntg(128)][kt(18)][lane*16] */
    for (size_t i = gid; i < (size_t)VV * 128 * KT_ENC * 512; i += stride) {
        int e = (int)(i & 15);
        int lane = (int)((i >> 4) & 31);
        size_t r = i >> 9;
        int kt = (int)(r % KT_ENC);
        size_t r2 = r / KT_ENC;
        int ntg = (int)(r2 & 127);
        int v = (int)(r2 >> 7);
        int K = kmap(e, lane >> 4);
        int n = ntg * 16 + (lane & 15);
        int k = kt * 32 + K;
        float val = (k < FF) ? W_ih[((size_t)v * G4H + n) * FF + k]
                             : W_hh[((size_t)v * G4H + n) * HH + (k - FF)];
        Wsw[i] = f2bf(val);
    }

    /* E) soft-attention weights -> swizzled bf16: [v][nt(32)][kt(32)][lane*16] */
    for (size_t i = gid; i < (size_t)VV * 32 * KT_SOFT * 512; i += stride) {
        int e = (int)(i & 15);
        int lane = (int)((i >> 4) & 31);
        size_t r = i >> 9;
        int kt = (int)(r & 31);
        size_t r2 = r >> 5;
        int nt = (int)(r2 & 31);
        int v = (int)(r2 >> 5);
        int K = kmap(e, lane >> 4);
        int n = nt * 16 + (lane & 15);
        int k = kt * 32 + K;
        softWs[i] = f2bf(soft_W[((size_t)v * HH + n) * (2 * HH) + k]);
    }

    /* F) hardwired weights (unnormalized) + per-step global max (values >= 0) */
    for (size_t i = gid; i < (size_t)TT * BB * VV * VV; i += stride) {
        int w_ = (int)(i & 15), vv = (int)((i >> 4) & 15);
        int b = (int)((i >> 8) & 31), t = (int)(i >> 13);
        size_t midx = (((size_t)(b * VV + vv) * TT) + t) * VV + w_;
        int r_ = rb[midx], cg = cog[midx];
        float m1 = mask[(b * VV + vv) * TT + t] * mask[(b * VV + w_) * TT + t];
        float val = fmaxf(dom[cg * NBINS + r_] - dist[midx], 0.f) * m1;
        wn[i] = val;
        atomicMax(maxi + t, __float_as_int(val));
    }
}

/* =================== init kernel 2: normalize w by per-step max ============ */
__global__ void k_norm(char* ws) {
    size_t gid = (size_t)blockIdx.x * blockDim.x + threadIdx.x;
    size_t stride = (size_t)gridDim.x * blockDim.x;
    float* wn = (float*)(ws + WS_WN);
    const int* maxi = (const int*)(ws + WS_MAX);
    for (size_t i = gid; i < (size_t)TT * BB * VV * VV; i += stride) {
        float mx = __int_as_float(maxi[i >> 13]);
        wn[i] = (mx > 0.f) ? wn[i] / mx : 0.f;
    }
}

/* =================== persistent fused model kernel ========================= */
/* grid = 64 WGs: (vessel v = bid>>2, h-slice quarter q = bid&3), 256 thr     */
__global__ __launch_bounds__(NTHR, 1) void vessel_main(
    char* ws, const float* seq_mask, const float* soft_b, const float* out_W,
    const float* out_b, float* out) {
    const u16*   seq_bf = (const u16*)(ws + WS_SEQBF);
    const u16*   xdec   = (const u16*)(ws + WS_XDEC);
    const u16*   Wsw    = (const u16*)(ws + WS_WSW);
    const u16*   softWs = (const u16*)(ws + WS_SOFTW);
    const float* bias   = (const float*)(ws + WS_BIAS);
    const float* wnorm  = (const float*)(ws + WS_WN);
    u16*   hbuf = (u16*)(ws + WS_HBUF);
    u16*   enc  = (u16*)(ws + WS_ENC);
    u16*   zbuf = (u16*)(ws + WS_ZBUF);
    float* h2   = (float*)(ws + WS_H2);
    unsigned* ctr = (unsigned*)(ws + WS_CTR);

    __shared__ alignas(32) u16 lds_a[2 * KT_SOFT * 32 * 16]; /* 64KB, A frags */
    __shared__ float lds_alpha[8][64];

    const int tid = threadIdx.x;
    const int wid = tid >> 5;
    const int lane = tid & 31;
    const int nl = lane & 15;
    const int lh = lane >> 4;
    const int bid = blockIdx.x;
    const int v = bid >> 2;
    const int q = bid & 3;

    unsigned nbar = 0;

    const u16* Wsw_v = Wsw + (size_t)v * (128 * KT_ENC * 512);
    const float* bias_v = bias + v * G4H;

    v8f cst[2];
    cst[0] = vsplat(0.f);
    cst[1] = vsplat(0.f);

    /* ---------------- encoder (t=0..63) + decoder (step 64) ---------------- */
    for (int step = 0; step <= TT; ++step) {
        const int t = (step < TT) ? step : (TT - 1);

        /* build swizzled A = [x_t ; wh] in LDS: [mt(2)][kt(18)][lane(32)][e(16)] */
        for (int idx = tid; idx < 2 * KT_ENC * 32 * 16; idx += NTHR) {
            int e = idx & 15;
            int ln = (idx >> 4) & 31;
            int kt = (idx >> 9) % KT_ENC;
            int mt = (idx >> 9) / KT_ENC;
            int K = kmap(e, ln >> 4);
            int b = mt * 16 + (ln & 15);
            int k = kt * 32 + K;
            float val;
            if (k < FF) {
                val = (step < TT)
                          ? bf2f(seq_bf[((size_t)(b * VV + v) * TT + step) * FF + k])
                          : bf2f(xdec[(b * VV + v) * FF + k]);
            } else {
                int h = k - FF;
                const float* wnp = wnorm + (((size_t)t * BB + b) * VV + v) * VV;
                const u16* hb = hbuf + (size_t)(b * VV) * HH + h;
                float a = 0.f;
#pragma unroll
                for (int w_ = 0; w_ < VV; ++w_) a += wnp[w_] * bf2f(hb[(size_t)w_ * HH]);
                val = a;
            }
            lds_a[idx] = f2bf(val);
        }
        __syncthreads();

        /* per-wave GEMM: n-tile (gate*32 + q*8 + wid), both m-tiles, 4 gates */
        v8f acc[4][2];
        const u16* bbase[4];
#pragma unroll
        for (int g = 0; g < 4; ++g) {
            float bv = bias_v[g * 512 + q * 128 + wid * 16 + nl];
            acc[g][0] = vsplat(bv);
            acc[g][1] = vsplat(bv);
            bbase[g] = Wsw_v + ((size_t)(g * 32 + q * 8 + wid) * KT_ENC) * 512 + lane * 16;
        }
        for (int kt = 0; kt < KT_ENC; ++kt) {
            v16bf a0 = *(const v16bf*)(lds_a + ((0 * KT_ENC + kt) * 32 + lane) * 16);
            v16bf a1 = *(const v16bf*)(lds_a + ((1 * KT_ENC + kt) * 32 + lane) * 16);
#pragma unroll
            for (int g = 0; g < 4; ++g) {
                const u16* bp = bbase[g] + (size_t)kt * 512;
                if (kt + 1 < KT_ENC) __builtin_prefetch(bp + 512, 0, 1);
                v16bf bm = *(const v16bf*)bp;
                acc[g][0] = __builtin_amdgcn_wmma_f32_16x16x32_bf16(
                    false, a0, false, bm, (short)0, acc[g][0], false, false);
                acc[g][1] = __builtin_amdgcn_wmma_f32_16x16x32_bf16(
                    false, a1, false, bm, (short)0, acc[g][1], false, false);
            }
        }

        /* gates -> cell/hidden update; c persists in VGPRs */
        const int hidx = q * 128 + wid * 16 + nl;
#pragma unroll
        for (int mt = 0; mt < 2; ++mt) {
#pragma unroll
            for (int r = 0; r < 8; ++r) {
                float ig = sigm(acc[0][mt][r]);
                float fg = sigm(acc[1][mt][r]);
                float gg = tanhf(acc[2][mt][r]);
                float og = sigm(acc[3][mt][r]);
                float c = fg * cst[mt][r] + ig * gg;
                cst[mt][r] = c;
                float hv = og * tanhf(c);
                int b = mt * 16 + r + 8 * lh;
                u16 h16 = f2bf(hv);
                hbuf[(size_t)(b * VV + v) * HH + hidx] = h16;
                if (step < TT)
                    enc[((size_t)(b * VV + v) * TT + step) * HH + hidx] = h16;
            }
        }
        grid_barrier(ctr, nbar);
    }

    /* ---------------- soft attention: scores, softmax, ctx, z -------------- */
    {
        int p = bid * 8 + wid; /* (b, vessel) pair: 512 total */
        int b = p >> 4, vv2 = p & 15;
        const u16* encb = enc + (size_t)(b * VV + vv2) * TT * HH;
        const u16* hb = hbuf + (size_t)(b * VV + vv2) * HH;
        const float* mk = seq_mask + (b * VV + vv2) * TT;
        float s0 = 0.f, s1 = 0.f;
        for (int h = 0; h < HH; ++h) {
            float hv = bf2f(hb[h]);
            s0 += bf2f(encb[(size_t)lane * HH + h]) * hv;
            s1 += bf2f(encb[(size_t)(lane + 32) * HH + h]) * hv;
        }
        s0 *= mk[lane];
        s1 *= mk[lane + 32];
        float mx = fmaxf(s0, s1);
        for (int off = 16; off; off >>= 1) mx = fmaxf(mx, __shfl_xor(mx, off, 32));
        float e0 = __expf(s0 - mx), e1 = __expf(s1 - mx);
        float sm = e0 + e1;
        for (int off = 16; off; off >>= 1) sm += __shfl_xor(sm, off, 32);
        float inv = 1.0f / sm;
        lds_alpha[wid][lane] = e0 * inv;
        lds_alpha[wid][lane + 32] = e1 * inv;
        __syncthreads();

        const float* al = lds_alpha[wid];
        size_t zrow_c = ((size_t)vv2 * 32 + (b >> 1)) * 1024 + (size_t)(b & 1) * 512;
        size_t zrow_h = ((size_t)vv2 * 32 + 16 + (b >> 1)) * 1024 + (size_t)(b & 1) * 512;
        for (int j = 0; j < 16; ++j) {
            int h = lane + 32 * j;
            float a = 0.f;
            for (int tt2 = 0; tt2 < TT; ++tt2)
                a += al[tt2] * bf2f(encb[(size_t)tt2 * HH + h]);
            zbuf[zrow_c + h] = f2bf(a); /* ctx rows 0..15 (torch view-reinterpret) */
            zbuf[zrow_h + h] = hb[h];   /* h   rows 16..31                        */
        }
    }
    grid_barrier(ctr, nbar);

    /* ---------------- per-vessel 2H->H tanh linear via WMMA ---------------- */
    {
        for (int idx = tid; idx < 2 * KT_SOFT * 32 * 16; idx += NTHR) {
            int e = idx & 15;
            int ln = (idx >> 4) & 31;
            int kt = (idx >> 9) & 31;
            int mt = idx >> 14;
            int K = kmap(e, ln >> 4);
            int bp2 = mt * 16 + (ln & 15);
            int k = kt * 32 + K;
            lds_a[idx] = zbuf[((size_t)v * 32 + bp2) * 1024 + k];
        }
        __syncthreads();

        int nt2 = q * 8 + wid;
        float bv = soft_b[v * HH + nt2 * 16 + nl];
        v8f a2[2];
        a2[0] = vsplat(bv);
        a2[1] = vsplat(bv);
        const u16* swb = softWs + ((size_t)(v * 32 + nt2) * KT_SOFT) * 512 + lane * 16;
        for (int kt = 0; kt < KT_SOFT; ++kt) {
            v16bf a0 = *(const v16bf*)(lds_a + ((0 * KT_SOFT + kt) * 32 + lane) * 16);
            v16bf a1 = *(const v16bf*)(lds_a + ((1 * KT_SOFT + kt) * 32 + lane) * 16);
            const u16* bp = swb + (size_t)kt * 512;
            if (kt + 1 < KT_SOFT) __builtin_prefetch(bp + 512, 0, 1);
            v16bf bm = *(const v16bf*)bp;
            a2[0] = __builtin_amdgcn_wmma_f32_16x16x32_bf16(false, a0, false, bm,
                                                            (short)0, a2[0], false, false);
            a2[1] = __builtin_amdgcn_wmma_f32_16x16x32_bf16(false, a1, false, bm,
                                                            (short)0, a2[1], false, false);
        }
        int n = nt2 * 16 + nl;
#pragma unroll
        for (int mt = 0; mt < 2; ++mt) {
#pragma unroll
            for (int r = 0; r < 8; ++r) {
                int bp2 = mt * 16 + r + 8 * lh;
                h2[(size_t)(bp2 * VV + v) * HH + n] = tanhf(a2[mt][r]);
            }
        }
    }
    grid_barrier(ctr, nbar);

    /* ---------------- output head: relu(h2 @ out_W^T + out_b)[:, :4] ------- */
    {
        int gt = bid * NTHR + tid;
        if (gt < BB * VV * OUTN) {
            int f = gt & 3, vv3 = (gt >> 2) & 15, b = gt >> 6;
            float a = out_b[f];
            const float* hp = h2 + (size_t)(b * VV + vv3) * HH;
            const float* wp = out_W + (size_t)f * HH;
            for (int h = 0; h < HH; ++h) a += hp[h] * wp[h];
            out[gt] = fmaxf(a, 0.f);
        }
    }
}

extern "C" void kernel_launch(void* const* d_in, const int* in_sizes, int n_in,
                              void* d_out, int out_size, void* d_ws, size_t ws_size,
                              hipStream_t stream) {
    const float* seq    = (const float*)d_in[0];
    const float* dist   = (const float*)d_in[1];
    const int*   rb     = (const int*)d_in[2];
    const int*   cog    = (const int*)d_in[3];
    const float* mask   = (const float*)d_in[4];
    const float* maxv   = (const float*)d_in[5];
    const float* minv   = (const float*)d_in[6];
    const float* W_ih   = (const float*)d_in[7];
    const float* W_hh   = (const float*)d_in[8];
    const float* b_ih   = (const float*)d_in[9];
    const float* b_hh   = (const float*)d_in[10];
    const float* dom    = (const float*)d_in[11];
    const float* soft_W = (const float*)d_in[12];
    const float* soft_b = (const float*)d_in[13];
    const float* out_W  = (const float*)d_in[14];
    const float* out_b  = (const float*)d_in[15];
    char* ws = (char*)d_ws;
    float* out = (float*)d_out;

    k_zero<<<1024, NTHR, 0, stream>>>(ws);
    k_prep<<<2048, NTHR, 0, stream>>>(ws, seq, dist, rb, cog, mask, maxv, minv,
                                      W_ih, W_hh, b_ih, b_hh, dom, soft_W);
    k_norm<<<512, NTHR, 0, stream>>>(ws);
    vessel_main<<<NWG, NTHR, 0, stream>>>(ws, mask, soft_b, out_W, out_b, out);
}
